// TSSCS_Dynamic_4561255268662
// MI455X (gfx1250) — matrix operand order repaired and checked by reading, not compiled, not run on hardware
//
#include <hip/hip_runtime.h>

// ---------------- problem constants ----------------
#define D_MODEL   640
#define D_INNER   1280
#define D_STATE   8
#define DT_RANK   40
#define K_CONV    4
#define BATCH     2
#define HH        24
#define WW        24
#define W6        (6 * WW)          // 144
#define SEQ_L     (HH * W6)         // 3456
#define M_TOK     (BATCH * SEQ_L)   // 6912
#define N_XZ      (2 * D_INNER)     // 2560
#define RK_PAD    64                // DT_RANK padded to 64 (2 K-steps of 32)
#define XP_NPAD   64                // xproj N (56) padded to 64
#define OUT_IMG   (BATCH * D_MODEL * HH * WW)  // 737280 per output map

#define BM 128                      // block tile M (8 waves x 16 rows)
#define BN 64                       // block tile N (4 wmma col-tiles)

typedef __attribute__((ext_vector_type(16))) __bf16 v16bf;
typedef __attribute__((ext_vector_type(8)))  float  v8f;
typedef __attribute__((ext_vector_type(4)))  unsigned int u32x4;
typedef __attribute__((ext_vector_type(8)))  int  i32x8;
typedef __attribute__((ext_vector_type(4)))  int  i32x4;

union FragU { uint4 q[2]; v16bf v; };

#if __has_builtin(__builtin_amdgcn_tensor_load_to_lds) && \
    __has_builtin(__builtin_amdgcn_s_wait_tensorcnt)
#define USE_TDM 1
#endif

__device__ __forceinline__ unsigned short f2bf(float f) {
    unsigned u = __float_as_uint(f);
    unsigned r = 0x7fffu + ((u >> 16) & 1u);
    return (unsigned short)((u + r) >> 16);
}

#ifdef USE_TDM
// ------------------------------------------------------------------
// Tensor Data Mover: 2-D tile (bf16) global -> LDS.
// D# per CDNA5 ISA 8.3/8.4: group0 {count,lds_addr,global_addr,type=2},
// group1 {data_size=2B, tensor_dim0/1 = tile dims, tile_dim0=32 (K),
// tile_dim1=rows, tensor_dim0_stride = lda}.  Groups 2/3 (and the
// trailing group of the 6-arg builtin) zero for a 2-D tensor.
// ------------------------------------------------------------------
__device__ __forceinline__ void tdm_load_tile(unsigned lds_addr,
                                              const unsigned short* gptr,
                                              unsigned rows,
                                              unsigned long long stride_elems) {
    unsigned long long ga = (unsigned long long)gptr;
    u32x4 g0;
    g0[0] = 1u;                                     // count=1 (valid), user mode
    g0[1] = lds_addr;                               // LDS byte address
    g0[2] = (unsigned)(ga & 0xFFFFFFFFu);           // global_addr[31:0]
    g0[3] = (unsigned)((ga >> 32) & 0x1FFFFFFu)     // global_addr[56:32]
          | (2u << 30);                             // type = 2 (image)
    i32x8 g1;
    g1[0] = (int)(1u << 16);                        // data_size = 1 -> 2 bytes
    g1[1] = (int)((32u & 0xFFFFu) << 16);           // tensor_dim0 = 32 (lo16)
    g1[2] = (int)((rows & 0xFFFFu) << 16);          // dim0 hi=0 | tensor_dim1 lo16
    g1[3] = (int)(32u << 16);                       // dim1 hi=0 | tile_dim0 = 32
    g1[4] = (int)rows;                              // tile_dim1 = rows, tile_dim2=0
    g1[5] = (int)(unsigned)(stride_elems & 0xFFFFFFFFull);   // dim0_stride lo32
    g1[6] = (int)(unsigned)((stride_elems >> 32) & 0xFFFFull); // dim0_stride hi16
    g1[7] = 0;
    i32x4 gz4 = {0, 0, 0, 0};
    i32x8 gz8 = {0, 0, 0, 0, 0, 0, 0, 0};
    __builtin_amdgcn_tensor_load_to_lds(g0, g1, gz4, gz4, gz8, 0);
}
#endif

__device__ __forceinline__ unsigned lds_off(const void* p) {
    return (unsigned)(unsigned long long)p;   // low 32 bits = LDS offset
}

// =====================================================================
// Weight convert + transpose: src [K][N] f32 -> dst [Np][Kp] bf16 (padded)
// =====================================================================
__global__ void convertT_kernel(const float* __restrict__ src,
                                unsigned short* __restrict__ dst,
                                int K, int N, int Kp, int Np) {
    int idx = blockIdx.x * blockDim.x + threadIdx.x;
    int total = Kp * Np;
    if (idx >= total) return;
    int n = idx / Kp;
    int k = idx - n * Kp;
    float v = (k < K && n < N) ? src[(long)k * N + n] : 0.0f;
    dst[(long)n * Kp + k] = f2bf(v);
}

// =====================================================================
// Gather six maps (concat along W, NCHW -> token-major) + LayerNorm.
// =====================================================================
__global__ void prep_kernel(const float* __restrict__ x1, const float* __restrict__ x2,
                            const float* __restrict__ x3, const float* __restrict__ x4,
                            const float* __restrict__ x5, const float* __restrict__ x6,
                            const float* __restrict__ ln_w, const float* __restrict__ ln_b,
                            float* __restrict__ seq, unsigned short* __restrict__ hbf) {
    __shared__ float vals[D_MODEL];
    __shared__ float red[256];
    __shared__ float red2[256];
    const int token = blockIdx.x;
    const int b  = token / SEQ_L;
    const int l  = token - b * SEQ_L;
    const int hh = l / W6;
    const int w6 = l - hh * W6;
    const int img = w6 / WW;
    const int w   = w6 - img * WW;
    const float* xs[6] = {x1, x2, x3, x4, x5, x6};
    const float* xp = xs[img];

    float s = 0.f, s2 = 0.f;
    for (int c = threadIdx.x; c < D_MODEL; c += 256) {
        float v = xp[(((long)b * D_MODEL + c) * HH + hh) * WW + w];
        vals[c] = v;
        s += v; s2 += v * v;
    }
    red[threadIdx.x] = s; red2[threadIdx.x] = s2;
    __syncthreads();
    for (int st = 128; st > 0; st >>= 1) {
        if (threadIdx.x < st) {
            red[threadIdx.x]  += red[threadIdx.x + st];
            red2[threadIdx.x] += red2[threadIdx.x + st];
        }
        __syncthreads();
    }
    const float mu  = red[0] * (1.0f / D_MODEL);
    const float var = red2[0] * (1.0f / D_MODEL) - mu * mu;
    const float inv = rsqrtf(var + 1e-5f);

    for (int c = threadIdx.x; c < D_MODEL; c += 256) {
        float v = vals[c];
        seq[(long)token * D_MODEL + c] = v;
        float hn = (v - mu) * inv * ln_w[c] + ln_b[c];
        hbf[(long)token * D_MODEL + c] = f2bf(hn);
    }
}

// =====================================================================
// One K-step of the 128x64 tile: A frag + all 4 B frags loaded first,
// then 4 back-to-back v_wmma (better XDL issue, fewer dscnt fences).
// Fragment layouts per CDNA5 ISA 7.12.2.
// =====================================================================
__device__ __forceinline__ void compute_tile(const unsigned short* As_,
                                             const unsigned short* Bs_,
                                             int wave, int lhalf, int l16,
                                             v8f c[4]) {
    FragU a;
    const unsigned short* arow = As_ + (wave * 16 + l16) * 32;
    a.q[0] = *(const uint4*)(arow + lhalf * 8);
    a.q[1] = *(const uint4*)(arow + lhalf * 8 + 16);
    FragU b[4];
#pragma unroll
    for (int nt = 0; nt < 4; ++nt) {
        const unsigned short* brow = Bs_ + (nt * 16 + l16) * 32;
        b[nt].q[0] = *(const uint4*)(brow + lhalf * 8);
        b[nt].q[1] = *(const uint4*)(brow + lhalf * 8 + 16);
    }
#pragma unroll
    for (int nt = 0; nt < 4; ++nt)
        c[nt] = __builtin_amdgcn_wmma_f32_16x16x32_bf16(
            false, a.v, false, b[nt].v, (short)0, c[nt], false, false);
}

// =====================================================================
// WMMA GEMM core: 128(M) x 64(N) block tile, 256 threads = 8 wave32.
// A: [M][lda] bf16 row-major, B: [N][ldb] bf16 (weights pre-transposed).
// TDM path: wave 0 drives double-buffered tensor_load_to_lds; waits on
// TENSORcnt; barriers publish LDS to the compute waves.
// =====================================================================
__device__ __forceinline__ void gemm_core_128x64(
        const unsigned short* __restrict__ Ag, int lda,
        const unsigned short* __restrict__ Bg, int ldb,
        int K, long mBase, long nBase, v8f c[4]) {
#ifdef USE_TDM
    __shared__ unsigned short As[2][BM * 32];
    __shared__ unsigned short Bs[2][BN * 32];
#else
    __shared__ unsigned short As[1][BM * 32];
    __shared__ unsigned short Bs[1][BN * 32];
#endif
    const int tid   = threadIdx.x;       // 0..255
    const int wave  = tid >> 5;          // 0..7 -> 16 M-rows each
    const int lane  = tid & 31;
    const int lhalf = lane >> 4;
    const int l16   = lane & 15;

#ifdef USE_TDM
    const int nT = K >> 5;
    const unsigned short* Abase = Ag + mBase * (long)lda;
    const unsigned short* Bbase = Bg + nBase * (long)ldb;
    if (wave == 0) {  // prologue: tile 0 -> buffer 0
        tdm_load_tile(lds_off(&As[0][0]), Abase, BM, (unsigned long long)lda);
        tdm_load_tile(lds_off(&Bs[0][0]), Bbase, BN, (unsigned long long)ldb);
    }
    for (int i = 0; i < nT; ++i) {
        const int buf = i & 1;
        if (wave == 0) {
            if (i + 1 < nT) {  // issue next tile into other buffer, then
                               // wait until only those 2 remain outstanding
                const int k1 = (i + 1) << 5;
                tdm_load_tile(lds_off(&As[buf ^ 1][0]), Abase + k1, BM,
                              (unsigned long long)lda);
                tdm_load_tile(lds_off(&Bs[buf ^ 1][0]), Bbase + k1, BN,
                              (unsigned long long)ldb);
                __builtin_amdgcn_s_wait_tensorcnt(2);
            } else {
                __builtin_amdgcn_s_wait_tensorcnt(0);
            }
        }
        __syncthreads();                       // tile i visible to all waves
        compute_tile(&As[buf][0], &Bs[buf][0], wave, lhalf, l16, c);
        __syncthreads();                       // done reading before overwrite
    }
#else
    for (int k0 = 0; k0 < K; k0 += 32) {
        __syncthreads();
        {   // A tile 128x32: 512 x 16B segments, 2 per thread (unrolled)
            int r = tid >> 2, sg = tid & 3;
            *(uint4*)(&As[0][0] + r * 32 + sg * 8) =
                *(const uint4*)(Ag + (mBase + r) * (long)lda + k0 + sg * 8);
            int idx2 = tid + 256;
            r = idx2 >> 2; sg = idx2 & 3;
            *(uint4*)(&As[0][0] + r * 32 + sg * 8) =
                *(const uint4*)(Ag + (mBase + r) * (long)lda + k0 + sg * 8);
            // B tile 64x32: 256 segments, 1 per thread
            r = tid >> 2; sg = tid & 3;
            *(uint4*)(&Bs[0][0] + r * 32 + sg * 8) =
                *(const uint4*)(Bg + (nBase + r) * (long)ldb + k0 + sg * 8);
        }
        if (k0 + 32 < K) {
            __builtin_prefetch(Ag + (mBase + tid) * (long)lda + k0 + 32, 0, 1);
            __builtin_prefetch(Bg + (nBase + (tid >> 2)) * (long)ldb + k0 + 32, 0, 1);
        }
        __syncthreads();
        compute_tile(&As[0][0], &Bs[0][0], wave, lhalf, l16, c);
    }
#endif
}

// accumulator element (nt, j) -> global (m, n):
//   m = mBase + wave*16 + lhalf*8 + j ;  n = nBase + nt*16 + l16
#define EPILOGUE_DECODE()                                   \
    const int tid   = threadIdx.x;                          \
    const int wave  = tid >> 5;                             \
    const int lane  = tid & 31;                             \
    const int lhalf = lane >> 4;                            \
    const int l16   = lane & 15;

// ---------------- GEMM 1: xz = h @ W_in + b_in ----------------
__global__ void __launch_bounds__(256)
gemm_xz_kernel(const unsigned short* __restrict__ hbf,
               const unsigned short* __restrict__ WinT,
               const float* __restrict__ b_in,
               float* __restrict__ u_pre, float* __restrict__ z) {
    const long nBase = (long)blockIdx.x * BN;
    const long mBase = (long)blockIdx.y * BM;
    v8f zero = {0, 0, 0, 0, 0, 0, 0, 0};
    v8f c[4] = {zero, zero, zero, zero};
    gemm_core_128x64(hbf, D_MODEL, WinT, D_MODEL, D_MODEL, mBase, nBase, c);
    EPILOGUE_DECODE();
#pragma unroll
    for (int nt = 0; nt < 4; ++nt) {
        long n = nBase + nt * 16 + l16;
        float bias = b_in[n];
#pragma unroll
        for (int j = 0; j < 8; ++j) {
            long m = mBase + wave * 16 + lhalf * 8 + j;
            float v = c[nt][j] + bias;
            if (n < D_INNER) u_pre[m * D_INNER + n] = v;
            else             z[m * D_INNER + (n - D_INNER)] = v;
        }
    }
}

// ---------------- causal depthwise conv (K=4) + SiLU ----------------
__global__ void conv_silu_kernel(const float* __restrict__ u_pre,
                                 const float* __restrict__ conv_w,
                                 const float* __restrict__ conv_b,
                                 float* __restrict__ u_conv,
                                 unsigned short* __restrict__ u_conv_bf) {
    long idx = (long)blockIdx.x * blockDim.x + threadIdx.x;
    if (idx >= (long)M_TOK * D_INNER) return;
    long m = idx / D_INNER;
    int  d = (int)(idx - m * D_INNER);
    int  b = (int)(m / SEQ_L);
    int  l = (int)(m - (long)b * SEQ_L);
    float acc = conv_b[d];
#pragma unroll
    for (int j = 0; j < K_CONV; ++j) {
        int t = l - (K_CONV - 1) + j;
        if (t >= 0)
            acc += conv_w[d * K_CONV + j] *
                   u_pre[((long)b * SEQ_L + t) * D_INNER + d];
    }
    float s = acc / (1.0f + expf(-acc));   // SiLU
    u_conv[idx]    = s;
    u_conv_bf[idx] = f2bf(s);
}

// ---------------- GEMM 2: proj = u @ W_xproj (N padded to 64) ---------
__global__ void __launch_bounds__(256)
gemm_xproj_kernel(const unsigned short* __restrict__ ubf,
                  const unsigned short* __restrict__ WxpT,
                  unsigned short* __restrict__ dtr_bf,  // [M][64] cols 40..63 zero
                  float* __restrict__ Bm, float* __restrict__ Cm) {
    const long mBase = (long)blockIdx.y * BM;
    v8f zero = {0, 0, 0, 0, 0, 0, 0, 0};
    v8f c[4] = {zero, zero, zero, zero};
    gemm_core_128x64(ubf, D_INNER, WxpT, D_INNER, D_INNER, mBase, 0, c);
    EPILOGUE_DECODE();
#pragma unroll
    for (int nt = 0; nt < 4; ++nt) {
        int n = nt * 16 + l16;
#pragma unroll
        for (int j = 0; j < 8; ++j) {
            long m = mBase + wave * 16 + lhalf * 8 + j;
            float v = c[nt][j];
            if (n < DT_RANK) {
                dtr_bf[m * RK_PAD + n] = f2bf(v);
            } else {
                dtr_bf[m * RK_PAD + n] = 0;  // zero K-padding for dt GEMM
                if (n < DT_RANK + D_STATE)          Bm[m * D_STATE + (n - DT_RANK)] = v;
                else if (n < DT_RANK + 2 * D_STATE) Cm[m * D_STATE + (n - DT_RANK - D_STATE)] = v;
            }
        }
    }
}

// ---------------- GEMM 3: dt = softplus(dt_r @ W_dt + b_dt) ------------
__global__ void __launch_bounds__(256)
gemm_dt_kernel(const unsigned short* __restrict__ dtr_bf,
               const unsigned short* __restrict__ WdtT,
               const float* __restrict__ b_dt,
               float* __restrict__ dt) {
    const long nBase = (long)blockIdx.x * BN;
    const long mBase = (long)blockIdx.y * BM;
    v8f zero = {0, 0, 0, 0, 0, 0, 0, 0};
    v8f c[4] = {zero, zero, zero, zero};
    gemm_core_128x64(dtr_bf, RK_PAD, WdtT, RK_PAD, RK_PAD, mBase, nBase, c);
    EPILOGUE_DECODE();
#pragma unroll
    for (int nt = 0; nt < 4; ++nt) {
        long n = nBase + nt * 16 + l16;
        float bias = b_dt[n];
#pragma unroll
        for (int j = 0; j < 8; ++j) {
            long m = mBase + wave * 16 + lhalf * 8 + j;
            float x = c[nt][j] + bias;
            float sp = (x > 20.f) ? x : log1pf(expf(x));   // softplus
            dt[m * D_INNER + n] = sp;
        }
    }
}

// ---------------- selective scan (fp32), fused SiLU(z) gate ------------
__global__ void scan_kernel(const float* __restrict__ dt,
                            const float* __restrict__ u,
                            const float* __restrict__ Bm,
                            const float* __restrict__ Cm,
                            const float* __restrict__ z,
                            const float* __restrict__ A_log,
                            const float* __restrict__ D_skip,
                            unsigned short* __restrict__ ygated_bf) {
    const int d = blockIdx.x * blockDim.x + threadIdx.x;   // 0..1279
    const int b = blockIdx.y;
    if (d >= D_INNER) return;
    float A[D_STATE], h[D_STATE];
#pragma unroll
    for (int s = 0; s < D_STATE; ++s) {
        A[s] = -expf(A_log[d * D_STATE + s]);
        h[s] = 0.f;
    }
    const float Dd = D_skip[d];
    const long base = (long)b * SEQ_L;
    for (int t = 0; t < SEQ_L; ++t) {
        const long row = base + t;
        const float dtv = dt[row * D_INNER + d];
        const float uv  = u[row * D_INNER + d];
        const float zv  = z[row * D_INNER + d];
        const float dtu = dtv * uv;
        const float* Bp = Bm + row * D_STATE;
        const float* Cp = Cm + row * D_STATE;
        float y = 0.f;
#pragma unroll
        for (int s = 0; s < D_STATE; ++s) {
            h[s] = expf(dtv * A[s]) * h[s] + dtu * Bp[s];
            y += h[s] * Cp[s];
        }
        y += uv * Dd;
        const float g = zv / (1.f + expf(-zv));            // SiLU gate
        ygated_bf[row * D_INNER + d] = f2bf(y * g);
    }
}

// ---------------- GEMM 4: out = seq + y @ W_out + b_out, scatter -------
__global__ void __launch_bounds__(256)
gemm_out_kernel(const unsigned short* __restrict__ ybf,
                const unsigned short* __restrict__ WoutT,
                const float* __restrict__ b_out,
                const float* __restrict__ seq,
                float* __restrict__ out) {
    const long nBase = (long)blockIdx.x * BN;
    const long mBase = (long)blockIdx.y * BM;
    v8f zero = {0, 0, 0, 0, 0, 0, 0, 0};
    v8f c[4] = {zero, zero, zero, zero};
    gemm_core_128x64(ybf, D_INNER, WoutT, D_INNER, D_INNER, mBase, nBase, c);
    EPILOGUE_DECODE();
#pragma unroll
    for (int nt = 0; nt < 4; ++nt) {
        long n = nBase + nt * 16 + l16;
        float bias = b_out[n];
#pragma unroll
        for (int j = 0; j < 8; ++j) {
            long m = mBase + wave * 16 + lhalf * 8 + j;
            float v = c[nt][j] + bias + seq[m * D_MODEL + n];
            int b  = (int)(m / SEQ_L);
            int l  = (int)(m - (long)b * SEQ_L);
            int hh = l / W6;
            int w6 = l - hh * W6;
            int img = w6 / WW;
            int w   = w6 - img * WW;
            out[(long)img * OUT_IMG +
                (((long)b * D_MODEL + n) * HH + hh) * WW + w] = v;
        }
    }
}

// =====================================================================
extern "C" void kernel_launch(void* const* d_in, const int* in_sizes, int n_in,
                              void* d_out, int out_size, void* d_ws, size_t ws_size,
                              hipStream_t stream) {
    const float* x1      = (const float*)d_in[0];
    const float* x2      = (const float*)d_in[1];
    const float* x3      = (const float*)d_in[2];
    const float* x4      = (const float*)d_in[3];
    const float* x5      = (const float*)d_in[4];
    const float* x6      = (const float*)d_in[5];
    const float* ln_w    = (const float*)d_in[6];
    const float* ln_b    = (const float*)d_in[7];
    const float* W_in    = (const float*)d_in[8];
    const float* b_in    = (const float*)d_in[9];
    const float* conv_w  = (const float*)d_in[10];
    const float* conv_b  = (const float*)d_in[11];
    const float* W_xproj = (const float*)d_in[12];
    const float* W_dt    = (const float*)d_in[13];
    const float* b_dt    = (const float*)d_in[14];
    const float* A_log   = (const float*)d_in[15];
    const float* D_skip  = (const float*)d_in[16];
    const float* W_out   = (const float*)d_in[17];
    const float* b_out   = (const float*)d_in[18];
    float* out = (float*)d_out;

    size_t off = 0;
    auto alloc = [&](size_t bytes) -> void* {
        void* p = (char*)d_ws + off;
        off += (bytes + 255) & ~(size_t)255;
        return p;
    };
    float*          seq    = (float*)         alloc((size_t)M_TOK * D_MODEL * 4);
    unsigned short* hbf    = (unsigned short*)alloc((size_t)M_TOK * D_MODEL * 2);
    unsigned short* WinT   = (unsigned short*)alloc((size_t)N_XZ * D_MODEL * 2);
    unsigned short* WxpT   = (unsigned short*)alloc((size_t)XP_NPAD * D_INNER * 2);
    unsigned short* WdtT   = (unsigned short*)alloc((size_t)D_INNER * RK_PAD * 2);
    unsigned short* WoutT  = (unsigned short*)alloc((size_t)D_MODEL * D_INNER * 2);
    float*          u_pre  = (float*)         alloc((size_t)M_TOK * D_INNER * 4);
    float*          zbuf   = (float*)         alloc((size_t)M_TOK * D_INNER * 4);
    float*          u_conv = (float*)         alloc((size_t)M_TOK * D_INNER * 4);
    unsigned short* ucbf   = (unsigned short*)alloc((size_t)M_TOK * D_INNER * 2);
    unsigned short* dtrbf  = (unsigned short*)alloc((size_t)M_TOK * RK_PAD * 2);
    float*          Bmb    = (float*)         alloc((size_t)M_TOK * D_STATE * 4);
    float*          Cmb    = (float*)         alloc((size_t)M_TOK * D_STATE * 4);
    float*          dtbuf  = (float*)         alloc((size_t)M_TOK * D_INNER * 4);
    unsigned short* ygbf   = (unsigned short*)alloc((size_t)M_TOK * D_INNER * 2);
    (void)ws_size; (void)in_sizes; (void)n_in; (void)out_size;

    // 1. weight convert/transpose to bf16 [N][Kp]
    {
        int t;
        t = N_XZ * D_MODEL;
        convertT_kernel<<<(t + 255) / 256, 256, 0, stream>>>(W_in, WinT, D_MODEL, N_XZ, D_MODEL, N_XZ);
        t = XP_NPAD * D_INNER;
        convertT_kernel<<<(t + 255) / 256, 256, 0, stream>>>(W_xproj, WxpT, D_INNER, DT_RANK + 2 * D_STATE, D_INNER, XP_NPAD);
        t = D_INNER * RK_PAD;
        convertT_kernel<<<(t + 255) / 256, 256, 0, stream>>>(W_dt, WdtT, DT_RANK, D_INNER, RK_PAD, D_INNER);
        t = D_MODEL * D_INNER;
        convertT_kernel<<<(t + 255) / 256, 256, 0, stream>>>(W_out, WoutT, D_INNER, D_MODEL, D_INNER, D_MODEL);
    }

    // 2. gather + layernorm
    prep_kernel<<<M_TOK, 256, 0, stream>>>(x1, x2, x3, x4, x5, x6, ln_w, ln_b, seq, hbf);

    // 3. in-projection GEMM (WMMA bf16, TDM-staged)
    gemm_xz_kernel<<<dim3(N_XZ / BN, M_TOK / BM), 256, 0, stream>>>(hbf, WinT, b_in, u_pre, zbuf);

    // 4. depthwise causal conv + SiLU
    {
        long t = (long)M_TOK * D_INNER;
        conv_silu_kernel<<<(unsigned)((t + 255) / 256), 256, 0, stream>>>(u_pre, conv_w, conv_b, u_conv, ucbf);
    }

    // 5. x-projection GEMM -> dt_r, B, C
    gemm_xproj_kernel<<<dim3(1, M_TOK / BM), 256, 0, stream>>>(ucbf, WxpT, dtrbf, Bmb, Cmb);

    // 6. dt GEMM + softplus
    gemm_dt_kernel<<<dim3(D_INNER / BN, M_TOK / BM), 256, 0, stream>>>(dtrbf, WdtT, b_dt, dtbuf);

    // 7. selective scan + SiLU(z) gate (fp32 recurrence)
    scan_kernel<<<dim3(D_INNER / 256, BATCH), 256, 0, stream>>>(dtbuf, u_conv, Bmb, Cmb, zbuf, A_log, D_skip, ygbf);

    // 8. out-projection GEMM + residual + scatter to 6 maps
    gemm_out_kernel<<<dim3(D_MODEL / BN, M_TOK / BM), 256, 0, stream>>>(ygbf, WoutT, b_out, seq, out);
}
